// DSRAModel_7834020348435
// MI455X (gfx1250) — matrix operand order repaired and checked by reading, not compiled, not run on hardware
//
#include <hip/hip_runtime.h>
#include <math.h>

typedef unsigned short u16;
typedef __attribute__((ext_vector_type(16))) __bf16 v16bf;
typedef __attribute__((ext_vector_type(8)))  float  v8f;

#define VOCAB_ 32000
#define DIM_   1024
#define K_     128
#define KR_    8
#define CHUNK_ 256
#define SEQ_   2048
#define BATCH_ 2
#define LAM_   0.9f

__device__ __forceinline__ u16 f2bf(float f) {
  union { float f; unsigned u; } v; v.f = f;
  unsigned r = v.u + 0x7FFFu + ((v.u >> 16) & 1u);   // round-to-nearest-even
  return (u16)(r >> 16);
}

// ---------------- f32 -> bf16 convert ----------------
__global__ void cvt_bf16_kernel(const float* __restrict__ in, u16* __restrict__ out, size_t n) {
  size_t i = (size_t)blockIdx.x * blockDim.x + threadIdx.x;
  size_t stride = (size_t)gridDim.x * blockDim.x;
  for (; i < n; i += stride) out[i] = f2bf(in[i]);
}

// ---------------- embedding gather + causal-context sum (LCTX=4) ----------------
__global__ void embed_ctx_kernel(const int* __restrict__ x, const float* __restrict__ emb,
                                 float* __restrict__ ctx, u16* __restrict__ ctxbf) {
  int s = blockIdx.x, b = blockIdx.y, t = threadIdx.x;
  float acc[4] = {0.f, 0.f, 0.f, 0.f};
  #pragma unroll
  for (int o = 0; o < 4; ++o) {
    int sp = s - o;
    if (sp >= 0) {
      int idx = x[b * SEQ_ + sp];
      const float* e = emb + (size_t)idx * DIM_;
      #pragma unroll
      for (int j = 0; j < 4; ++j) acc[j] += e[t + j * 256];
    }
  }
  size_t base = ((size_t)b * SEQ_ + s) * DIM_;
  #pragma unroll
  for (int j = 0; j < 4; ++j) {
    ctx[base + t + j * 256]   = acc[j];
    ctxbf[base + t + j * 256] = f2bf(acc[j]);
  }
}

// ---------------- tiled bf16 WMMA GEMM (double-buffered, async A staging) -----
// C[M,N] = epi( alpha * A[M,K] @ B[K,N] + (Cin?) + (bias?) )
// Block tile 128x128, BK=32; 8 waves, each wave 32x64 (2x4 wmma frags).
// A tile staged with global_load_async_to_lds_b128 (ASYNCcnt), B tile staged
// through VGPRs with a transpose scatter. One barrier per K-tile.
// epi: 0=none, 1=elu+1 feature map, 2=causal tril mask (zero where col>row)
__global__ __launch_bounds__(256) void gemm_bf16_kernel(
    const u16* __restrict__ A, long long sAb, int lda,
    const u16* __restrict__ B, long long sBb, int ldb,
    const float* __restrict__ Cin, long long sCinb,
    float* __restrict__ Cout, long long sCoutb, int ldc,
    u16* __restrict__ Obf, long long sObfb, int ldobf,
    u16* __restrict__ ObfT, long long sObfTb, int ldobft,
    const float* __restrict__ bias,
    int M, int N, int Kd, float alpha, int epi)
{
  __shared__ __align__(16) u16 sA[2][128][40];   // row-major A tile, padded pitch
  __shared__ __align__(16) u16 sBt[2][128][40];  // transposed B tile [n][k], padded

  const int tid   = threadIdx.x;
  const int lane  = tid & 31;
  const int wave  = tid >> 5;
  const int waveM = wave >> 1;          // 0..3
  const int waveN = wave & 1;           // 0..1
  const int r     = lane & 15;
  const int hh    = lane >> 4;

  const int n0 = blockIdx.x * 128;
  const int m0 = blockIdx.y * 128;
  const int bz = blockIdx.z;

  const u16* Ab = A + (long long)bz * sAb;
  const u16* Bb = B + (long long)bz * sBb;

  // per-thread staging coordinates (512 x 16B chunks per tile, 2 per thread)
  int arow[2], acol[2], bk[2], bn[2];
  #pragma unroll
  for (int i = 0; i < 2; ++i) {
    int ci = tid + i * 256;
    arow[i] = ci >> 2;  acol[i] = (ci & 3) * 8;
    bk[i]   = ci >> 4;  bn[i]   = (ci & 15) * 8;
  }

  v8f acc[2][4];
  #pragma unroll
  for (int i = 0; i < 2; ++i)
    #pragma unroll
    for (int j = 0; j < 4; ++j)
      #pragma unroll
      for (int e = 0; e < 8; ++e) acc[i][j][e] = 0.f;

  union FragU { uint4 q[2]; v16bf v; };

  uint4 breg[2];

  auto issueA = [&](int k0, int buf) {
    #pragma unroll
    for (int i = 0; i < 2; ++i) {
      // low 32 bits of the generic shared address == LDS byte offset (LDS aperture)
      unsigned lo = (unsigned)(uintptr_t)(&sA[buf][arow[i]][acol[i]]);
      const u16* gp = Ab + (long long)(m0 + arow[i]) * lda + (k0 + acol[i]);
      asm volatile("global_load_async_to_lds_b128 %0, %1, off"
                   :: "v"(lo), "v"((unsigned long long)(uintptr_t)gp)
                   : "memory");
    }
  };
  auto issueB = [&](int k0) {
    #pragma unroll
    for (int i = 0; i < 2; ++i)
      breg[i] = *(const uint4*)(Bb + (long long)(k0 + bk[i]) * ldb + (n0 + bn[i]));
  };
  auto commitB = [&](int buf) {
    #pragma unroll
    for (int i = 0; i < 2; ++i) {
      union { uint4 v; u16 s[8]; } t;
      t.v = breg[i];
      #pragma unroll
      for (int e = 0; e < 8; ++e) sBt[buf][bn[i] + e][bk[i]] = t.s[e];
    }
  };

  // ---- prologue: stage tile 0 into buffer 0 ----
  issueA(0, 0);
  issueB(0);
  commitB(0);
  asm volatile("s_wait_asynccnt 0x0" ::: "memory");
  __syncthreads();

  const int T = Kd >> 5;
  int buf = 0;
  for (int t = 0; t < T; ++t) {
    const bool more = (t + 1) < T;
    const int k1 = (t + 1) << 5;
    if (more) {           // issue next tile early: async A -> LDS, B -> VGPRs
      issueA(k1, buf ^ 1);
      issueB(k1);
    }

    FragU afr[2], bfr[4];
    #pragma unroll
    for (int fm = 0; fm < 2; ++fm) {
      int row = waveM * 32 + fm * 16 + r;
      afr[fm].q[0] = *(const uint4*)(&sA[buf][row][hh * 8]);        // K = h*8 .. +7
      afr[fm].q[1] = *(const uint4*)(&sA[buf][row][16 + hh * 8]);   // K = 16+h*8 ..
    }
    #pragma unroll
    for (int fn = 0; fn < 4; ++fn) {
      int col = waveN * 64 + fn * 16 + r;
      bfr[fn].q[0] = *(const uint4*)(&sBt[buf][col][hh * 8]);       // K = h*8 .. +7
      bfr[fn].q[1] = *(const uint4*)(&sBt[buf][col][16 + hh * 8]);  // K = 16+h*8 ..
    }
    #pragma unroll
    for (int fm = 0; fm < 2; ++fm)
      #pragma unroll
      for (int fn = 0; fn < 4; ++fn)
        acc[fm][fn] = __builtin_amdgcn_wmma_f32_16x16x32_bf16(
            false, afr[fm].v, false, bfr[fn].v, (short)0, acc[fm][fn], false, false);

    if (more) {
      commitB(buf ^ 1);   // implicit s_wait_loadcnt here, after the WMMAs
      asm volatile("s_wait_asynccnt 0x0" ::: "memory");
    }
    __syncthreads();      // emits s_wait_dscnt + barrier
    buf ^= 1;
  }

  const float* CinB  = Cin  ? Cin  + (long long)bz * sCinb  : (const float*)0;
  float*       CoutB = Cout ? Cout + (long long)bz * sCoutb : (float*)0;
  u16*         ObfB  = Obf  ? Obf  + (long long)bz * sObfb  : (u16*)0;
  u16*         ObfTB = ObfT ? ObfT + (long long)bz * sObfTb : (u16*)0;

  #pragma unroll
  for (int fm = 0; fm < 2; ++fm) {
    #pragma unroll
    for (int fn = 0; fn < 4; ++fn) {
      int gcol  = n0 + waveN * 64 + fn * 16 + r;
      int growb = m0 + waveM * 32 + fm * 16 + hh * 8;
      #pragma unroll
      for (int v = 0; v < 8; ++v) {
        int grow = growb + v;
        float val = acc[fm][fn][v] * alpha;
        if (CinB) val += CinB[(long long)grow * ldc + gcol];
        if (bias) val += bias[gcol];
        if (epi == 1)      val = (val > 0.f) ? (val + 1.f) : __expf(val);  // elu(x)+1
        else if (epi == 2) { if (gcol > grow) val = 0.f; }                 // causal mask
        if (CoutB) CoutB[(long long)grow * ldc   + gcol] = val;
        if (ObfB)  ObfB [(long long)grow * ldobf + gcol] = f2bf(val);
        if (ObfTB) ObfTB[(long long)gcol * ldobft + grow] = f2bf(val);
      }
    }
  }
}

// ---------------- h2 = out1 + byp@Vb + St  (bf16 out) ----------------
__global__ void bypass_add_kernel(const float* __restrict__ out1, const float* __restrict__ byp,
                                  const float* __restrict__ Vb, const float* __restrict__ St,
                                  u16* __restrict__ h2) {
  int i = blockIdx.x, b = blockIdx.y, t = threadIdx.x;
  float bp[KR_];
  #pragma unroll
  for (int q = 0; q < KR_; ++q) bp[q] = byp[b * KR_ + q];
  size_t base = ((size_t)b * CHUNK_ + i) * DIM_;
  #pragma unroll
  for (int j = 0; j < 4; ++j) {
    int d = t + j * 256;
    float acc = out1[base + d] + St[b * DIM_ + d];
    #pragma unroll
    for (int q = 0; q < KR_; ++q) acc += bp[q] * Vb[q * DIM_ + d];
    h2[base + d] = f2bf(acc);
  }
}

// ---------------- xmean over chunk + EMA time-state update ----------------
__global__ void xmean_st_kernel(const float* __restrict__ ctx, float* __restrict__ xmean,
                                float* __restrict__ St, int cbase) {
  int b = blockIdx.y;
  int d = blockIdx.x * 256 + threadIdx.x;
  const float* p = ctx + ((size_t)b * SEQ_ + cbase) * DIM_ + d;
  float s = 0.f;
  for (int i = 0; i < CHUNK_; ++i) s += p[(size_t)i * DIM_];
  s *= (1.f / (float)CHUNK_);
  xmean[b * DIM_ + d] = s;
  St[b * DIM_ + d] = LAM_ * St[b * DIM_ + d] + (1.f - LAM_) * s;
}

// ---------------- byp = xmean @ Ub  (1024 x 8) ----------------
__global__ void byp_kernel(const float* __restrict__ xmean, const float* __restrict__ Ub,
                           float* __restrict__ byp) {
  int b = blockIdx.x, t = threadIdx.x;
  float p[KR_];
  #pragma unroll
  for (int q = 0; q < KR_; ++q) p[q] = 0.f;
  for (int d = t; d < DIM_; d += 256) {
    float xm = xmean[b * DIM_ + d];
    #pragma unroll
    for (int q = 0; q < KR_; ++q) p[q] += xm * Ub[d * KR_ + q];
  }
  __shared__ float red[KR_][256];
  #pragma unroll
  for (int q = 0; q < KR_; ++q) red[q][t] = p[q];
  __syncthreads();
  for (int s2 = 128; s2 > 0; s2 >>= 1) {
    if (t < s2) {
      #pragma unroll
      for (int q = 0; q < KR_; ++q) red[q][t] += red[q][t + s2];
    }
    __syncthreads();
  }
  if (t < KR_) byp[b * KR_ + t] = red[t][0];
}

// ---------------- LayerNorm -> bf16 ----------------
__global__ void layernorm_kernel(const float* __restrict__ h, const float* __restrict__ g,
                                 const float* __restrict__ bb, u16* __restrict__ hn) {
  size_t row = blockIdx.x;
  int t = threadIdx.x;
  const float* p = h + row * DIM_;
  float v0[4], s = 0.f, sq = 0.f;
  #pragma unroll
  for (int j = 0; j < 4; ++j) {
    v0[j] = p[t + j * 256];
    s += v0[j]; sq += v0[j] * v0[j];
  }
  __shared__ float rs[256], rq[256];
  rs[t] = s; rq[t] = sq;
  __syncthreads();
  for (int s2 = 128; s2 > 0; s2 >>= 1) {
    if (t < s2) { rs[t] += rs[t + s2]; rq[t] += rq[t + s2]; }
    __syncthreads();
  }
  float mu  = rs[0] * (1.f / (float)DIM_);
  float var = rq[0] * (1.f / (float)DIM_) - mu * mu;
  float inv = rsqrtf(var + 1e-5f);
  #pragma unroll
  for (int j = 0; j < 4; ++j) {
    int d = t + j * 256;
    hn[row * DIM_ + d] = f2bf((v0[j] - mu) * inv * g[d] + bb[d]);
  }
}

// ---------------- host-side GEMM launch helper ----------------
static inline void gemm_launch(hipStream_t st,
    const u16* A, long long sA, int lda,
    const u16* B, long long sB, int ldb,
    const float* Cin, long long sCin,
    float* Cout, long long sCout, int ldc,
    u16* Obf, long long sObf, int ldobf,
    u16* ObfT, long long sObfT, int ldobft,
    const float* bias, int M, int N, int Kd, float alpha, int epi, int batch)
{
  dim3 grid(N / 128, M / 128, batch), block(256);
  gemm_bf16_kernel<<<grid, block, 0, st>>>(A, sA, lda, B, sB, ldb, Cin, sCin,
                                           Cout, sCout, ldc, Obf, sObf, ldobf,
                                           ObfT, sObfT, ldobft, bias, M, N, Kd, alpha, epi);
}

extern "C" void kernel_launch(void* const* d_in, const int* in_sizes, int n_in,
                              void* d_out, int out_size, void* d_ws, size_t ws_size,
                              hipStream_t stream)
{
  (void)in_sizes; (void)n_in; (void)out_size; (void)ws_size;
  const int*   x    = (const int*)  d_in[0];
  const float* emb  = (const float*)d_in[1];
  const float* Wq   = (const float*)d_in[2];
  const float* Wk   = (const float*)d_in[3];
  const float* Wv   = (const float*)d_in[4];
  const float* Wo   = (const float*)d_in[5];
  const float* Ub   = (const float*)d_in[6];
  const float* Vb   = (const float*)d_in[7];
  const float* ln_g = (const float*)d_in[8];
  const float* ln_b = (const float*)d_in[9];
  const float* Wout = (const float*)d_in[10];
  const float* bout = (const float*)d_in[11];
  float* out = (float*)d_out;

  char* ws = (char*)d_ws;
  size_t off = 0;
  auto alloc = [&](size_t bytes) -> void* {
    off = (off + 255) & ~(size_t)255;
    void* p = ws + off;
    off += bytes;
    return p;
  };

  const size_t BSD = (size_t)BATCH_ * SEQ_ * DIM_;          // 4,194,304
  float* ctx_f   = (float*)alloc(BSD * 4);
  u16*   ctx_bf  = (u16*)  alloc(BSD * 2);
  float* h_f     = (float*)alloc(BSD * 4);
  u16*   hn_bf   = (u16*)  alloc(BSD * 2);
  u16*   wq_bf   = (u16*)  alloc((size_t)DIM_ * K_ * 2);
  u16*   wk_bf   = (u16*)  alloc((size_t)DIM_ * K_ * 2);
  u16*   wv_bf   = (u16*)  alloc((size_t)DIM_ * DIM_ * 2);
  u16*   wo_bf   = (u16*)  alloc((size_t)DIM_ * DIM_ * 2);
  u16*   wout_bf = (u16*)  alloc((size_t)DIM_ * VOCAB_ * 2);
  u16*   q_bf    = (u16*)  alloc((size_t)BATCH_ * CHUNK_ * K_ * 2);
  u16*   k_bf    = (u16*)  alloc((size_t)BATCH_ * CHUNK_ * K_ * 2);
  u16*   kT_bf   = (u16*)  alloc((size_t)BATCH_ * K_ * CHUNK_ * 2);
  float* v_f     = (float*)alloc((size_t)BATCH_ * CHUNK_ * DIM_ * 4);
  u16*   v_bf    = (u16*)  alloc((size_t)BATCH_ * CHUNK_ * DIM_ * 2);
  u16*   attn_bf = (u16*)  alloc((size_t)BATCH_ * CHUNK_ * CHUNK_ * 2);
  float* out1_f  = (float*)alloc((size_t)BATCH_ * CHUNK_ * DIM_ * 4);
  u16*   h2_bf   = (u16*)  alloc((size_t)BATCH_ * CHUNK_ * DIM_ * 2);
  u16*   u_bf    = (u16*)  alloc((size_t)BATCH_ * CHUNK_ * DIM_ * 2);
  float* Sst     = (float*)alloc((size_t)BATCH_ * K_ * DIM_ * 4);
  u16*   Sst_bf  = (u16*)  alloc((size_t)BATCH_ * K_ * DIM_ * 2);
  float* byp     = (float*)alloc((size_t)BATCH_ * KR_ * 4);
  float* St      = (float*)alloc((size_t)BATCH_ * DIM_ * 4);
  float* xmean   = (float*)alloc((size_t)BATCH_ * DIM_ * 4);

  auto cvt = [&](const float* src, u16* dst, size_t n) {
    size_t blocks = (n + 2047) / 2048;
    if (blocks > 4096) blocks = 4096;
    cvt_bf16_kernel<<<dim3((unsigned)blocks), 256, 0, stream>>>(src, dst, n);
  };

  // weights -> bf16
  cvt(Wq,   wq_bf,   (size_t)DIM_ * K_);
  cvt(Wk,   wk_bf,   (size_t)DIM_ * K_);
  cvt(Wv,   wv_bf,   (size_t)DIM_ * DIM_);
  cvt(Wo,   wo_bf,   (size_t)DIM_ * DIM_);
  cvt(Wout, wout_bf, (size_t)DIM_ * VOCAB_);

  // embedding + causal context
  embed_ctx_kernel<<<dim3(SEQ_, BATCH_), 256, 0, stream>>>(x, emb, ctx_f, ctx_bf);

  // zero carried state (deterministic per call)
  hipMemsetAsync(Sst,    0, (size_t)BATCH_ * K_ * DIM_ * 4, stream);
  hipMemsetAsync(Sst_bf, 0, (size_t)BATCH_ * K_ * DIM_ * 2, stream);
  hipMemsetAsync(byp,    0, (size_t)BATCH_ * KR_ * 4, stream);
  hipMemsetAsync(St,     0, (size_t)BATCH_ * DIM_ * 4, stream);

  const float scale = 0.08838834764831845f;  // 1/sqrt(K)
  const long long ctxStride = (long long)SEQ_ * DIM_;

  for (int c = 0; c < SEQ_ / CHUNK_; ++c) {
    const u16*   cc_bf = ctx_bf + (size_t)c * CHUNK_ * DIM_;
    // 1. q = phi(ctx_c @ Wq)
    gemm_launch(stream, cc_bf, ctxStride, DIM_, wq_bf, 0, K_,
                0, 0, 0, 0, 0, q_bf, (long long)CHUNK_ * K_, K_, 0, 0, 0,
                0, CHUNK_, K_, DIM_, 1.f, 1, BATCH_);
    // 2. k = phi(ctx_c @ Wk)  (also write kT)
    gemm_launch(stream, cc_bf, ctxStride, DIM_, wk_bf, 0, K_,
                0, 0, 0, 0, 0, k_bf, (long long)CHUNK_ * K_, K_,
                kT_bf, (long long)K_ * CHUNK_, CHUNK_,
                0, CHUNK_, K_, DIM_, 1.f, 1, BATCH_);
    // 3. v = ctx_c @ Wv   (f32 + bf16)
    gemm_launch(stream, cc_bf, ctxStride, DIM_, wv_bf, 0, DIM_,
                0, 0, v_f, (long long)CHUNK_ * DIM_, DIM_,
                v_bf, (long long)CHUNK_ * DIM_, DIM_, 0, 0, 0,
                0, CHUNK_, DIM_, DIM_, 1.f, 0, BATCH_);
    // 4. attn = tril(scale * q @ kT)
    gemm_launch(stream, q_bf, (long long)CHUNK_ * K_, K_,
                kT_bf, (long long)K_ * CHUNK_, CHUNK_,
                0, 0, 0, 0, 0,
                attn_bf, (long long)CHUNK_ * CHUNK_, CHUNK_, 0, 0, 0,
                0, CHUNK_, CHUNK_, K_, scale, 2, BATCH_);
    // 5. out1 = scale * (q @ Sst)
    gemm_launch(stream, q_bf, (long long)CHUNK_ * K_, K_,
                Sst_bf, (long long)K_ * DIM_, DIM_,
                0, 0, out1_f, (long long)CHUNK_ * DIM_, DIM_,
                0, 0, 0, 0, 0, 0,
                0, CHUNK_, DIM_, K_, scale, 0, BATCH_);
    // 6. out1 += attn @ v
    gemm_launch(stream, attn_bf, (long long)CHUNK_ * CHUNK_, CHUNK_,
                v_bf, (long long)CHUNK_ * DIM_, DIM_,
                out1_f, (long long)CHUNK_ * DIM_,
                out1_f, (long long)CHUNK_ * DIM_, DIM_,
                0, 0, 0, 0, 0, 0,
                0, CHUNK_, DIM_, CHUNK_, 1.f, 0, BATCH_);
    // 7. h2 = out1 + byp@Vb + St   (uses previous-chunk byp/St)
    bypass_add_kernel<<<dim3(CHUNK_, BATCH_), 256, 0, stream>>>(out1_f, byp, Vb, St, h2_bf);
    // 8. h[:, c] = h2 @ Wo
    gemm_launch(stream, h2_bf, (long long)CHUNK_ * DIM_, DIM_, wo_bf, 0, DIM_,
                0, 0, h_f + (size_t)c * CHUNK_ * DIM_, ctxStride, DIM_,
                0, 0, 0, 0, 0, 0,
                0, CHUNK_, DIM_, DIM_, 1.f, 0, BATCH_);
    // 9. u = v - scale * (k @ Sst)
    gemm_launch(stream, k_bf, (long long)CHUNK_ * K_, K_,
                Sst_bf, (long long)K_ * DIM_, DIM_,
                v_f, (long long)CHUNK_ * DIM_,
                0, 0, DIM_,
                u_bf, (long long)CHUNK_ * DIM_, DIM_, 0, 0, 0,
                0, CHUNK_, DIM_, K_, -scale, 0, BATCH_);
    // 10. Sst += kT @ u
    gemm_launch(stream, kT_bf, (long long)K_ * CHUNK_, CHUNK_,
                u_bf, (long long)CHUNK_ * DIM_, DIM_,
                Sst, (long long)K_ * DIM_,
                Sst, (long long)K_ * DIM_, DIM_,
                0, 0, 0, 0, 0, 0,
                0, K_, DIM_, CHUNK_, 1.f, 0, BATCH_);
    // 11. xmean / EMA St / byp / refresh Sst_bf
    xmean_st_kernel<<<dim3(DIM_ / 256, BATCH_), 256, 0, stream>>>(ctx_f, xmean, St, c * CHUNK_);
    byp_kernel<<<dim3(BATCH_), 256, 0, stream>>>(xmean, Ub, byp);
    cvt(Sst, Sst_bf, (size_t)BATCH_ * K_ * DIM_);
  }

  // LayerNorm -> bf16
  layernorm_kernel<<<dim3(BATCH_ * SEQ_), 256, 0, stream>>>(h_f, ln_g, ln_b, hn_bf);

  // logits = hn @ Wout + bout   (dominant GEMM: 4096 x 32000 x 1024)
  gemm_launch(stream, hn_bf, 0, DIM_, wout_bf, 0, VOCAB_,
              0, 0, out, 0, VOCAB_,
              0, 0, 0, 0, 0, 0,
              bout, BATCH_ * SEQ_, VOCAB_, DIM_, 1.f, 0, 1);
}
